// MoEWrapper_45578192945391
// MI455X (gfx1250) — matrix operand order repaired and checked by reading, not compiled, not run on hardware
//
#include <hip/hip_runtime.h>

#define HIDDEN 1024
#define FFN    4096
#define NEXP   8
#define NTOK   8192           // 2*4096 tokens
#define TOTROWS 16384         // NTOK * TOP_K(2)

typedef __attribute__((ext_vector_type(16))) __bf16          v16bf;
typedef __attribute__((ext_vector_type(8)))  float           v8f;
typedef __attribute__((ext_vector_type(4)))  float           f4;
typedef __attribute__((ext_vector_type(4)))  unsigned short  us4;
typedef int v4i_g __attribute__((vector_size(16)));   // matches builtin param type

struct F4x2 { f4 lo, hi; };

// ---- async global->LDS copy (CDNA5) with safe fallback --------------------
#if defined(__has_builtin)
#if __has_builtin(__builtin_amdgcn_global_load_async_to_lds_b128)
#define HAS_ASYNC_LDS 1
#endif
#endif
#ifndef HAS_ASYNC_LDS
#define HAS_ASYNC_LDS 0
#endif

static __device__ __forceinline__ void cp16(unsigned short* lds_dst,
                                            const unsigned short* gsrc) {
#if HAS_ASYNC_LDS
  __builtin_amdgcn_global_load_async_to_lds_b128(
      (__attribute__((address_space(1))) v4i_g*)(unsigned long long)gsrc,
      (__attribute__((address_space(3))) v4i_g*)(unsigned int)(unsigned long long)lds_dst,
      0, 0);
#else
  *(f4*)lds_dst = *(const f4*)gsrc;
#endif
}

static __device__ __forceinline__ void wait_async0() {
#if HAS_ASYNC_LDS
#if defined(__has_builtin) && __has_builtin(__builtin_amdgcn_s_wait_asynccnt)
  __builtin_amdgcn_s_wait_asynccnt(0);
#else
  asm volatile("s_wait_asynccnt 0x0" ::: "memory");
#endif
#endif
}

// ---- scheduler hint: cluster DS reads, then the WMMA burst ----------------
#if defined(__has_builtin)
#if __has_builtin(__builtin_amdgcn_sched_group_barrier)
#define SCHED_DS_THEN_WMMA(nds, nw)                           \
  do {                                                        \
    __builtin_amdgcn_sched_group_barrier(0x100, (nds), 0);    \
    __builtin_amdgcn_sched_group_barrier(0x008, (nw), 0);     \
  } while (0)
#endif
#endif
#ifndef SCHED_DS_THEN_WMMA
#define SCHED_DS_THEN_WMMA(nds, nw) do {} while (0)
#endif

// ---------------------------------------------------------------------------
static __device__ __forceinline__ unsigned short f2bf(float f) {
  unsigned u = __builtin_bit_cast(unsigned, f);
  u += 0x7FFFu + ((u >> 16) & 1u);          // round-to-nearest-even
  return (unsigned short)(u >> 16);
}

static __device__ __forceinline__ v8f wmma_bf16(v16bf a, v16bf b, v8f c) {
  return __builtin_amdgcn_wmma_f32_16x16x32_bf16(false, a, false, b, (short)0, c,
                                                 false, false);
}

// A fragment: 16x32 bf16 tile at LDS row m0 (row-major M x K, ld=40 ushorts).
static __device__ __forceinline__ v16bf frag_a(const unsigned short* As, int m0,
                                               int lane) {
  const unsigned short* row = As + (m0 + (lane & 15)) * 40 + ((lane >> 4) << 3);
  F4x2 t;
  t.lo = *(const f4*)(row);
  t.hi = *(const f4*)(row + 16);
  return __builtin_bit_cast(v16bf, t);
}

// B fragment from transposed LDS (N x K row-major, ld=40 ushorts).
static __device__ __forceinline__ v16bf frag_bt(const unsigned short* Bs, int n0,
                                                int lane) {
  const unsigned short* row = Bs + (n0 + (lane & 15)) * 40 + ((lane >> 4) << 3);
  F4x2 t;
  t.lo = *(const f4*)(row);
  t.hi = *(const f4*)(row + 16);
  return __builtin_bit_cast(v16bf, t);
}

// ---------------------------------------------------------------------------
__global__ __launch_bounds__(256) void moe_zero(float* __restrict__ out,
                                                int* __restrict__ cnt) {
  size_t i = (size_t)blockIdx.x * 256 + threadIdx.x;
  f4 z = {0.f, 0.f, 0.f, 0.f};
  ((f4*)out)[i] = z;
  if (blockIdx.x == 0 && threadIdx.x < NEXP) cnt[threadIdx.x] = 0;
}

// ---------------------------------------------------------------------------
__global__ __launch_bounds__(256) void moe_cvt(const float* __restrict__ src,
                                               unsigned short* __restrict__ dst) {
  size_t i = (size_t)blockIdx.x * 256 + threadIdx.x;
  f4 v = ((const f4*)src)[i];
  us4 o = { f2bf(v.x), f2bf(v.y), f2bf(v.z), f2bf(v.w) };
  ((us4*)dst)[i] = o;
}

// ---------------------------------------------------------------------------
// Transpose + convert: src fp32 [R,C] row-major -> dst bf16 [C,R] row-major.
__global__ __launch_bounds__(256) void moe_tcvt(const float* __restrict__ src0,
                                                unsigned short* __restrict__ dst0,
                                                int R, int C) {
  const size_t mat = (size_t)R * C;
  const float* src = src0 + (size_t)blockIdx.z * mat;
  unsigned short* dst = dst0 + (size_t)blockIdx.z * mat;
  const int cb = blockIdx.x * 64, rb = blockIdx.y * 64;

  __shared__ __align__(16) unsigned short T[64][80];   // T[c][r]
  const int tid = threadIdx.x;
#pragma unroll
  for (int it = 0; it < 4; ++it) {
    int idx = tid + it * 256;
    int r = idx >> 4, c4 = (idx & 15) << 2;
    f4 v = *(const f4*)(src + (size_t)(rb + r) * C + cb + c4);
    T[c4 + 0][r] = f2bf(v.x); T[c4 + 1][r] = f2bf(v.y);
    T[c4 + 2][r] = f2bf(v.z); T[c4 + 3][r] = f2bf(v.w);
  }
  __syncthreads();
#pragma unroll
  for (int it = 0; it < 2; ++it) {
    int idx = tid + it * 256;
    int c = idx >> 3, seg = (idx & 7) << 3;
    *(f4*)(dst + (size_t)(cb + c) * R + rb + seg) = *(const f4*)(&T[c][seg]);
  }
}

// ---------------------------------------------------------------------------
__global__ __launch_bounds__(256) void moe_router(
    const float* __restrict__ x, const float* __restrict__ gate_w,
    float* __restrict__ logits, float* __restrict__ combine,
    int* __restrict__ cnt, int* __restrict__ list) {
  const int lane = threadIdx.x & 31;
  const int wave = threadIdx.x >> 5;
  const int t = blockIdx.x * 8 + wave;                 // one wave32 per token

  float acc[NEXP];
#pragma unroll
  for (int e = 0; e < NEXP; ++e) acc[e] = 0.f;

  const float* xr = x + (size_t)t * HIDDEN;
  for (int k = lane; k < HIDDEN; k += 32) {
    float xv = xr[k];
    const float* gw = gate_w + k * NEXP;
#pragma unroll
    for (int e = 0; e < NEXP; ++e) acc[e] += xv * gw[e];
  }
#pragma unroll
  for (int e = 0; e < NEXP; ++e)
#pragma unroll
    for (int s = 16; s > 0; s >>= 1) acc[e] += __shfl_xor(acc[e], s, 32);

#pragma unroll
  for (int e = 0; e < NEXP; ++e)
    if (lane == e) logits[(size_t)t * NEXP + e] = acc[e];

  if (lane == 0) {
    float mx = acc[0];
#pragma unroll
    for (int e = 1; e < NEXP; ++e) mx = fmaxf(mx, acc[e]);
    float p[NEXP], sum = 0.f;
#pragma unroll
    for (int e = 0; e < NEXP; ++e) { p[e] = __expf(acc[e] - mx); sum += p[e]; }
#pragma unroll
    for (int e = 0; e < NEXP; ++e) p[e] /= sum;
    int i0 = 0;
#pragma unroll
    for (int e = 1; e < NEXP; ++e) if (p[e] > p[i0]) i0 = e;
    int i1 = (i0 == 0) ? 1 : 0;
#pragma unroll
    for (int e = 0; e < NEXP; ++e) if (e != i0 && p[e] > p[i1]) i1 = e;
    float s2 = p[i0] + p[i1];
    float w0 = p[i0] / s2, w1v = p[i1] / s2;
#pragma unroll
    for (int e = 0; e < NEXP; ++e) combine[(size_t)t * NEXP + e] = 0.f;
    combine[(size_t)t * NEXP + i0] = w0;
    combine[(size_t)t * NEXP + i1] = w1v;
    int p0 = atomicAdd(&cnt[i0], 1); list[i0 * NTOK + p0] = t;
    int p1 = atomicAdd(&cnt[i1], 1); list[i1 * NTOK + p1] = t;
  }
}

// ---------------------------------------------------------------------------
__global__ void moe_offsets(const int* __restrict__ cnt, int* __restrict__ offs) {
  if (threadIdx.x == 0 && blockIdx.x == 0) {
    int a = 0;
    for (int e = 0; e < NEXP; ++e) { offs[e] = a; a += cnt[e]; }
    offs[NEXP] = a;
  }
}

// ---------------------------------------------------------------------------
// h = silu(x@w1) * (x@w3), gathered rows, 128x64 tile, double-buffered LDS.
__global__ __launch_bounds__(256) void moe_gemm1(
    const unsigned short* __restrict__ xb, const unsigned short* __restrict__ w1t,
    const unsigned short* __restrict__ w3t, const int* __restrict__ cnt,
    const int* __restrict__ offs, const int* __restrict__ list,
    unsigned short* __restrict__ h_all) {
  const int e = blockIdx.z;
  const int rows = cnt[e];
  const int row0 = blockIdx.y * 128;
  if (row0 >= rows) return;
  const int c0 = blockIdx.x * 64;

  __shared__ __align__(16) unsigned short As[2][128][40];
  __shared__ __align__(16) unsigned short B1[2][64][40];
  __shared__ __align__(16) unsigned short B3[2][64][40];
  __shared__ int toks[128];

  const int tid = threadIdx.x, lane = tid & 31, wave = tid >> 5;
  if (tid < 128) {
    int r = row0 + tid;
    toks[tid] = (r < rows) ? list[e * NTOK + r] : 0;
  }
  __syncthreads();                                     // toks ready

  const unsigned short* w1e = w1t + (size_t)e * FFN * HIDDEN;
  const unsigned short* w3e = w3t + (size_t)e * FFN * HIDDEN;

  const int ar0 = tid >> 2, ar1 = ar0 + 64, aseg = (tid & 3) << 3;
  const int bn = tid >> 2, bseg = (tid & 3) << 3;
  const unsigned short* asrc0 = xb + (size_t)toks[ar0] * HIDDEN + aseg;
  const unsigned short* asrc1 = xb + (size_t)toks[ar1] * HIDDEN + aseg;
  const size_t boff = (size_t)(c0 + bn) * HIDDEN + bseg;

  auto stage = [&](int buf, int k0) {
    cp16(&As[buf][ar0][aseg], asrc0 + k0);
    cp16(&As[buf][ar1][aseg], asrc1 + k0);
    cp16(&B1[buf][bn][bseg], w1e + boff + k0);
    cp16(&B3[buf][bn][bseg], w3e + boff + k0);
  };

  v8f accG[4] = {}, accU[4] = {};
  stage(0, 0);

  for (int k0 = 0; k0 < HIDDEN; k0 += 32) {
    const int buf = (k0 >> 5) & 1;
    wait_async0();
    __syncthreads();
    if (k0 + 32 < HIDDEN) stage(buf ^ 1, k0 + 32);     // overlap with compute

    v16bf a = frag_a(&As[buf][0][0], wave * 16, lane); // one A feeds 8 WMMAs
    v16bf fb1[4], fb3[4];
#pragma unroll
    for (int j = 0; j < 4; ++j) {
      fb1[j] = frag_bt(&B1[buf][0][0], j * 16, lane);
      fb3[j] = frag_bt(&B3[buf][0][0], j * 16, lane);
    }
#pragma unroll
    for (int j = 0; j < 4; ++j) {
      accG[j] = wmma_bf16(a, fb1[j], accG[j]);
      accU[j] = wmma_bf16(a, fb3[j], accU[j]);
    }
    SCHED_DS_THEN_WMMA(18, 8);   // 2 A + 16 B ds_load_b128, then 8 WMMAs
  }

  const int rowoff = offs[e];
  const int mbase = wave * 16 + ((lane >> 4) << 3);
#pragma unroll
  for (int j = 0; j < 4; ++j) {
    int colg = c0 + j * 16 + (lane & 15);
#pragma unroll
    for (int r = 0; r < 8; ++r) {
      int gr = row0 + mbase + r;
      if (gr < rows) {
        float gv = accG[j][r];
        float hv = (gv / (1.0f + __expf(-gv))) * accU[j][r];
        h_all[(size_t)(rowoff + gr) * FFN + colg] = f2bf(hv);
      }
    }
  }
}

// ---------------------------------------------------------------------------
// y = h @ w2[e]. w2t: bf16 transposed [NEXP][HIDDEN][FFN].
__global__ __launch_bounds__(256) void moe_gemm2(
    const unsigned short* __restrict__ h_all, const unsigned short* __restrict__ w2t,
    const int* __restrict__ cnt, const int* __restrict__ offs,
    unsigned short* __restrict__ y_all) {
  const int e = blockIdx.z;
  const int rows = cnt[e];
  const int row0 = blockIdx.y * 128;
  if (row0 >= rows) return;
  const int c0 = blockIdx.x * 64;
  const int rowoff = offs[e];

  __shared__ __align__(16) unsigned short As[2][128][40];
  __shared__ __align__(16) unsigned short Bs[2][64][40];

  const int tid = threadIdx.x, lane = tid & 31, wave = tid >> 5;
  const unsigned short* w2e = w2t + (size_t)e * HIDDEN * FFN;

  const int ar0 = tid >> 2, aseg = (tid & 3) << 3;
  size_t arow0 = (size_t)(rowoff + row0 + ar0);
  size_t arow1 = arow0 + 64;
  if (arow0 > TOTROWS - 1) arow0 = TOTROWS - 1;        // clamp OOB scratch read
  if (arow1 > TOTROWS - 1) arow1 = TOTROWS - 1;
  const unsigned short* asrc0 = h_all + arow0 * FFN + aseg;
  const unsigned short* asrc1 = h_all + arow1 * FFN + aseg;
  const int bn = tid >> 2, bseg = (tid & 3) << 3;
  const unsigned short* bsrc = w2e + (size_t)(c0 + bn) * FFN + bseg;

  auto stage = [&](int buf, int k0) {
    cp16(&As[buf][ar0][aseg], asrc0 + k0);
    cp16(&As[buf][ar0 + 64][aseg], asrc1 + k0);
    cp16(&Bs[buf][bn][bseg], bsrc + k0);
  };

  v8f acc[4] = {};
  stage(0, 0);

  for (int k0 = 0; k0 < FFN; k0 += 32) {
    const int buf = (k0 >> 5) & 1;
    wait_async0();
    __syncthreads();
    if (k0 + 32 < FFN) stage(buf ^ 1, k0 + 32);

    v16bf a = frag_a(&As[buf][0][0], wave * 16, lane);
    v16bf fb[4];
#pragma unroll
    for (int j = 0; j < 4; ++j) fb[j] = frag_bt(&Bs[buf][0][0], j * 16, lane);
#pragma unroll
    for (int j = 0; j < 4; ++j) acc[j] = wmma_bf16(a, fb[j], acc[j]);
    SCHED_DS_THEN_WMMA(10, 4);   // 2 A + 8 B ds_load_b128, then 4 WMMAs
  }

  const int mbase = wave * 16 + ((lane >> 4) << 3);
#pragma unroll
  for (int j = 0; j < 4; ++j) {
    int colg = c0 + j * 16 + (lane & 15);
#pragma unroll
    for (int r = 0; r < 8; ++r) {
      int gr = row0 + mbase + r;
      if (gr < rows)
        y_all[(size_t)(rowoff + gr) * HIDDEN + colg] = f2bf(acc[j][r]);
    }
  }
}

// ---------------------------------------------------------------------------
// z = y @ unmerge_slice; out[tok] += z * combine[tok,e] (atomic f32).
__global__ __launch_bounds__(256) void moe_gemm3(
    const unsigned short* __restrict__ y_all, const unsigned short* __restrict__ umt,
    const int* __restrict__ cnt, const int* __restrict__ offs,
    const int* __restrict__ list, const float* __restrict__ combine,
    float* __restrict__ out) {
  const int e = blockIdx.z;
  const int rows = cnt[e];
  const int row0 = blockIdx.y * 128;
  if (row0 >= rows) return;
  const int c0 = blockIdx.x * 64;
  const int rowoff = offs[e];
  const int g = e >> 2, gi = e & 3;                    // GROUP_SIZE = 4

  __shared__ __align__(16) unsigned short As[2][128][40];
  __shared__ __align__(16) unsigned short Bs[2][64][40];

  const int tid = threadIdx.x, lane = tid & 31, wave = tid >> 5;
  const unsigned short* ube =
      umt + ((size_t)g * 4 * HIDDEN + (size_t)gi * HIDDEN) * HIDDEN;

  const int ar0 = tid >> 2, aseg = (tid & 3) << 3;
  size_t arow0 = (size_t)(rowoff + row0 + ar0);
  size_t arow1 = arow0 + 64;
  if (arow0 > TOTROWS - 1) arow0 = TOTROWS - 1;
  if (arow1 > TOTROWS - 1) arow1 = TOTROWS - 1;
  const unsigned short* asrc0 = y_all + arow0 * HIDDEN + aseg;
  const unsigned short* asrc1 = y_all + arow1 * HIDDEN + aseg;
  const int bn = tid >> 2, bseg = (tid & 3) << 3;
  const unsigned short* bsrc = ube + (size_t)(c0 + bn) * HIDDEN + bseg;

  auto stage = [&](int buf, int k0) {
    cp16(&As[buf][ar0][aseg], asrc0 + k0);
    cp16(&As[buf][ar0 + 64][aseg], asrc1 + k0);
    cp16(&Bs[buf][bn][bseg], bsrc + k0);
  };

  v8f acc[4] = {};
  stage(0, 0);

  for (int k0 = 0; k0 < HIDDEN; k0 += 32) {
    const int buf = (k0 >> 5) & 1;
    wait_async0();
    __syncthreads();
    if (k0 + 32 < HIDDEN) stage(buf ^ 1, k0 + 32);

    v16bf a = frag_a(&As[buf][0][0], wave * 16, lane);
    v16bf fb[4];
#pragma unroll
    for (int j = 0; j < 4; ++j) fb[j] = frag_bt(&Bs[buf][0][0], j * 16, lane);
#pragma unroll
    for (int j = 0; j < 4; ++j) acc[j] = wmma_bf16(a, fb[j], acc[j]);
    SCHED_DS_THEN_WMMA(10, 4);
  }

  const int mbase = wave * 16 + ((lane >> 4) << 3);
#pragma unroll
  for (int j = 0; j < 4; ++j) {
    int colg = c0 + j * 16 + (lane & 15);
#pragma unroll
    for (int r = 0; r < 8; ++r) {
      int gr = row0 + mbase + r;
      if (gr < rows) {
        int tok = list[e * NTOK + gr];
        float w = combine[(size_t)tok * NEXP + e];
        atomicAdd(&out[(size_t)tok * HIDDEN + colg], acc[j][r] * w);
      }
    }
  }
}

// ---------------------------------------------------------------------------
extern "C" void kernel_launch(void* const* d_in, const int* in_sizes, int n_in,
                              void* d_out, int out_size, void* d_ws, size_t ws_size,
                              hipStream_t stream) {
  (void)in_sizes; (void)n_in; (void)out_size; (void)ws_size;
  const float* x  = (const float*)d_in[0];
  const float* gw = (const float*)d_in[1];
  const float* w1 = (const float*)d_in[2];
  const float* w3 = (const float*)d_in[3];
  const float* w2 = (const float*)d_in[4];
  const float* um = (const float*)d_in[5];

  float* out    = (float*)d_out;
  float* logits = out + (size_t)NTOK * HIDDEN;

  const size_t MB = 1ull << 20;
  char* ws = (char*)d_ws;
  float* combine = (float*)ws;
  int* cnt  = (int*)(ws + 1 * MB);
  int* offs = (int*)(ws + 1 * MB + 64);
  int* list = (int*)(ws + 1 * MB + 128);
  unsigned short* xb  = (unsigned short*)(ws +   2 * MB);        // 16 MB
  unsigned short* w1t = (unsigned short*)(ws +  18 * MB);        // 64 MB
  unsigned short* w3t = (unsigned short*)(ws +  82 * MB);        // 64 MB
  unsigned short* w2t = (unsigned short*)(ws + 146 * MB);        // 64 MB
  unsigned short* umt = (unsigned short*)(ws + 210 * MB);        // 16 MB
  unsigned short* h_all = (unsigned short*)(ws + 226 * MB);      // 128 MB
  unsigned short* y_all = (unsigned short*)(ws + 354 * MB);      // 32 MB

  moe_zero<<<8192, 256, 0, stream>>>(out, cnt);
  moe_cvt<<<8192, 256, 0, stream>>>(x, xb);
  moe_tcvt<<<dim3(64, 16, NEXP), 256, 0, stream>>>(w1, w1t, HIDDEN, FFN);
  moe_tcvt<<<dim3(64, 16, NEXP), 256, 0, stream>>>(w3, w3t, HIDDEN, FFN);
  moe_tcvt<<<dim3(16, 64, NEXP), 256, 0, stream>>>(w2, w2t, FFN, HIDDEN);
  moe_tcvt<<<dim3(64, 16, 2),    256, 0, stream>>>(um, umt, HIDDEN, FFN);
  moe_router<<<NTOK / 8, 256, 0, stream>>>(x, gw, logits, combine, cnt, list);
  moe_offsets<<<1, 32, 0, stream>>>(cnt, offs);
  moe_gemm1<<<dim3(FFN / 64, NTOK / 128, NEXP), 256, 0, stream>>>(
      xb, w1t, w3t, cnt, offs, list, h_all);
  moe_gemm2<<<dim3(HIDDEN / 64, NTOK / 128, NEXP), 256, 0, stream>>>(
      h_all, w2t, cnt, offs, y_all);
  moe_gemm3<<<dim3(HIDDEN / 64, NTOK / 128, NEXP), 256, 0, stream>>>(
      y_all, umt, cnt, offs, list, combine, out);
}
